// Attention_55774445306282
// MI455X (gfx1250) — matrix operand order repaired and checked by reading, not compiled, not run on hardware
//
#include <hip/hip_runtime.h>
#include <math.h>

typedef __bf16 bf16;
typedef __bf16 v16bf __attribute__((ext_vector_type(16)));
typedef __bf16 v8bf  __attribute__((ext_vector_type(8)));
typedef float  v8f   __attribute__((ext_vector_type(8)));

#define DIMD   1024
#define HEADS  16
#define DH     64
#define SEQ    2048
#define BATCH  4
#define NTOK   (BATCH * SEQ)   // 8192 tokens

// -------- small helpers --------
__device__ __forceinline__ v16bf mk16(v8bf lo, v8bf hi) {
  v16bf r;
#pragma unroll
  for (int e = 0; e < 8; ++e) { r[e] = lo[e]; r[e + 8] = hi[e]; }
  return r;
}

__device__ __forceinline__ v8f zero8() {
  v8f z;
#pragma unroll
  for (int e = 0; e < 8; ++e) z[e] = 0.0f;
  return z;
}

// -------- fp32 -> bf16 conversion --------
__global__ void cvt_bf16_kernel(const float* __restrict__ in, bf16* __restrict__ out, int n) {
  int i = blockIdx.x * blockDim.x + threadIdx.x;
  int stride = gridDim.x * blockDim.x;
  for (; i < n; i += stride) out[i] = (bf16)in[i];
}

// -------- WMMA GEMM: C[M,N] = A[M,K] @ W[K,N] + bias --------
// Tile: 128(M) x 64(N), K-step 32, 8 waves (4x2), each wave 32x32 (2x2 wmma tiles).
constexpr int BM = 128, BN = 64, BK = 32;
constexpr int LDA = BK + 8;   // 40 bf16 -> 80B rows (16B aligned)
constexpr int LDB = BK + 8;   // Bs stored transposed [n][k]

__device__ __forceinline__ void stage_tiles(
    const bf16* __restrict__ A, const bf16* __restrict__ W,
    bf16* As, bf16* Bs, int rowBase, int colBase, int kt, int K, int N,
    int sar, int sak, int sbk, int sbn) {
  // A tile: 128 rows x 32 k, thread -> (row = t>>1, kchunk = (t&1)*16)
  const bf16* asrc = A + (size_t)(rowBase + sar) * K + kt * BK + sak;
  uint4 a0 = ((const uint4*)asrc)[0];
  uint4 a1 = ((const uint4*)asrc)[1];
  uint4* adst = (uint4*)&As[sar * LDA + sak];
  adst[0] = a0; adst[1] = a1;
  // B tile: 32 k x 64 n, stored transposed into Bs[n][k]
  const bf16* bsrc = W + (size_t)(kt * BK + sbk) * N + colBase + sbn;
  union { uint4 u; bf16 h[8]; } ub;
  ub.u = *(const uint4*)bsrc;
#pragma unroll
  for (int q = 0; q < 8; ++q) Bs[(sbn + q) * LDB + sbk] = ub.h[q];
}

// MODE 0: plain f32 output.  MODE 1: Q -> [b,h,n,dh] bf16.
// MODE 2: KV -> K [b,h,n,dh] bf16 (out0), V^T [b,h,dh,n] bf16 (out1).
template <int MODE>
__global__ __launch_bounds__(256) void gemm_wmma_kernel(
    const bf16* __restrict__ A, const bf16* __restrict__ W,
    const float* __restrict__ bias,
    void* __restrict__ out0, void* __restrict__ out1,
    int N, int K) {
  __shared__ bf16 As[2][BM * LDA];
  __shared__ bf16 Bs[2][BN * LDB];

  const int tid  = threadIdx.x;
  const int lane = tid & 31;
  const int wid  = tid >> 5;
  const int wm   = wid >> 1;     // 0..3: 32-row slice
  const int wn   = wid & 1;      // 0..1: 32-col slice
  const int lr   = lane & 15;
  const int hl   = lane >> 4;

  const int rowBase = blockIdx.x * BM;
  const int colBase = blockIdx.y * BN;

  const int sar = tid >> 1;
  const int sak = (tid & 1) * 16;
  const int sbk = tid >> 3;
  const int sbn = (tid & 7) * 8;

  v8f acc[2][2];
#pragma unroll
  for (int i = 0; i < 2; ++i)
#pragma unroll
    for (int j = 0; j < 2; ++j) acc[i][j] = zero8();

  const int NK = K / BK;
  stage_tiles(A, W, As[0], Bs[0], rowBase, colBase, 0, K, N, sar, sak, sbk, sbn);
  __syncthreads();

  for (int kt = 0; kt < NK; ++kt) {
    const int cur = kt & 1;
    if (kt + 1 < NK)
      stage_tiles(A, W, As[cur ^ 1], Bs[cur ^ 1], rowBase, colBase, kt + 1, K, N,
                  sar, sak, sbk, sbn);

    v16bf afr[2], bfr[2];
    const int ah  = hl * 8;    // A: lanes16-31 take K 8..15 / 24..31
    const int bh2 = hl * 8;    // B: lanes16-31 take K 8..15 / 24..31
#pragma unroll
    for (int i = 0; i < 2; ++i) {
      const bf16* p = &As[cur][(wm * 32 + i * 16 + lr) * LDA + ah];
      afr[i] = mk16(*(const v8bf*)p, *(const v8bf*)(p + 16));
    }
#pragma unroll
    for (int j = 0; j < 2; ++j) {
      const bf16* p = &Bs[cur][(wn * 32 + j * 16 + lr) * LDB + bh2];
      bfr[j] = mk16(*(const v8bf*)p, *(const v8bf*)(p + 16));
    }
#pragma unroll
    for (int i = 0; i < 2; ++i)
#pragma unroll
      for (int j = 0; j < 2; ++j)
        acc[i][j] = __builtin_amdgcn_wmma_f32_16x16x32_bf16(
            false, afr[i], false, bfr[j], (short)0, acc[i][j], false, false);
    __syncthreads();
  }

  // epilogue
#pragma unroll
  for (int i = 0; i < 2; ++i) {
#pragma unroll
    for (int j = 0; j < 2; ++j) {
      const int col = colBase + wn * 32 + j * 16 + lr;
      const float bv = bias[col];
#pragma unroll
      for (int r = 0; r < 8; ++r) {
        const int grow = rowBase + wm * 32 + i * 16 + r + hl * 8;
        const float v = acc[i][j][r] + bv;
        if (MODE == 0) {
          ((float*)out0)[(size_t)grow * N + col] = v;
        } else {
          const int bi  = grow >> 11;          // / SEQ
          const int tok = grow & (SEQ - 1);
          if (MODE == 1) {
            const int head = col >> 6, dd = col & 63;
            ((bf16*)out0)[(((size_t)(bi * HEADS + head) * SEQ + tok) << 6) + dd] = (bf16)v;
          } else {  // MODE == 2
            if (col < DIMD) {
              const int head = col >> 6, dd = col & 63;
              ((bf16*)out0)[(((size_t)(bi * HEADS + head) * SEQ + tok) << 6) + dd] = (bf16)v;
            } else {
              const int c2 = col - DIMD;
              const int head = c2 >> 6, dd = c2 & 63;
              ((bf16*)out1)[((size_t)(bi * HEADS + head) * DH + dd) * SEQ + tok] = (bf16)v;
            }
          }
        }
      }
    }
  }
}

// -------- Flash attention (causal, online softmax), WMMA bf16 --------
// Grid: (b*h, n/128). 8 waves/WG, each wave owns 16 query rows.
__global__ __launch_bounds__(256) void flash_attn_kernel(
    const bf16* __restrict__ Q,   // [b*h, n, dh]
    const bf16* __restrict__ Kb,  // [b*h, n, dh]
    const bf16* __restrict__ Vt,  // [b*h, dh, n]  (transposed)
    bf16* __restrict__ O) {       // [b, n, d]
  constexpr int BQ = 128, BKV = 64;
  constexpr int LDK = DH  + 8;   // Ks[key][dh]
  constexpr int LDV = BKV + 8;   // Vs[dh][key]
  constexpr int LDP = BKV + 8;   // Ps[row][key]
  __shared__ bf16 Ks[BKV * LDK];
  __shared__ bf16 Vs[DH * LDV];
  __shared__ bf16 Ps[8][16 * LDP];

  const int bh   = blockIdx.x;
  const int qb   = blockIdx.y;
  const int tid  = threadIdx.x;
  const int lane = tid & 31;
  const int wid  = tid >> 5;
  const int lr   = lane & 15;
  const int hl   = lane >> 4;

  const size_t base = (size_t)bh * SEQ * DH;
  const int qr0 = qb * BQ + wid * 16;
  const float scale = 0.125f;   // dh^-0.5

  // Q A-fragments: dh split into two K=32 chunks
  v16bf qa[2];
  {
    const bf16* qp = Q + base + (size_t)(qr0 + lr) * DH + hl * 8;
#pragma unroll
    for (int kk = 0; kk < 2; ++kk) {
      const bf16* p = qp + kk * 32;
      qa[kk] = mk16(*(const v8bf*)p, *(const v8bf*)(p + 16));
    }
  }

  v8f o[4];
#pragma unroll
  for (int jj = 0; jj < 4; ++jj) o[jj] = zero8();
  float m[8], l[8];
#pragma unroll
  for (int r = 0; r < 8; ++r) { m[r] = -__builtin_inff(); l[r] = 0.0f; }

  const int sky = tid >> 2;          // 0..63
  const int skd = (tid & 3) * 16;    // 0,16,32,48

  const int nkb = qb * 2 + 2;        // causal: only key blocks up to the diagonal
  for (int kbi = 0; kbi < nkb; ++kbi) {
    const int kb = kbi * BKV;
    __syncthreads();
    {  // K tile -> Ks[key][dh]
      const bf16* src = Kb + base + (size_t)(kb + sky) * DH + skd;
      uint4 x0 = ((const uint4*)src)[0], x1 = ((const uint4*)src)[1];
      uint4* dst = (uint4*)&Ks[sky * LDK + skd];
      dst[0] = x0; dst[1] = x1;
    }
    {  // V^T tile -> Vs[dh][key]
      const bf16* src = Vt + base + (size_t)sky * SEQ + kb + skd;
      uint4 x0 = ((const uint4*)src)[0], x1 = ((const uint4*)src)[1];
      uint4* dst = (uint4*)&Vs[sky * LDV + skd];
      dst[0] = x0; dst[1] = x1;
    }
    __syncthreads();

    // S = Q @ K^T over 4 key sub-tiles of 16
    v8f s[4];
#pragma unroll
    for (int j = 0; j < 4; ++j) {
      v8f sj = zero8();
#pragma unroll
      for (int kk = 0; kk < 2; ++kk) {
        const bf16* p = &Ks[(j * 16 + lr) * LDK + kk * 32 + hl * 8];
        v16bf kf = mk16(*(const v8bf*)p, *(const v8bf*)(p + 16));
        sj = __builtin_amdgcn_wmma_f32_16x16x32_bf16(
            false, qa[kk], false, kf, (short)0, sj, false, false);
      }
      s[j] = sj;
    }

    // online softmax (row = r + hl*8 within wave; reduce across 16-lane halves)
#pragma unroll
    for (int r = 0; r < 8; ++r) {
      const int qrow = qr0 + r + hl * 8;
      float rmax = -__builtin_inff();
#pragma unroll
      for (int j = 0; j < 4; ++j) {
        const int key = kb + j * 16 + lr;
        float v = s[j][r] * scale;
        v = (key > qrow) ? -__builtin_inff() : v;
        s[j][r] = v;
        rmax = fmaxf(rmax, v);
      }
#pragma unroll
      for (int off = 8; off >= 1; off >>= 1)
        rmax = fmaxf(rmax, __shfl_xor(rmax, off, 32));
      const float mn = fmaxf(m[r], rmax);
      const float alpha = __expf(fminf(m[r] - mn, 0.0f));  // robust vs -inf - -inf
      m[r] = mn;
      l[r] *= alpha;
#pragma unroll
      for (int jj = 0; jj < 4; ++jj) o[jj][r] *= alpha;
      float rsum = 0.0f;
#pragma unroll
      for (int j = 0; j < 4; ++j) {
        const float p = __expf(s[j][r] - mn);
        rsum += p;
        Ps[wid][(r + hl * 8) * LDP + j * 16 + lr] = (bf16)p;
      }
#pragma unroll
      for (int off = 8; off >= 1; off >>= 1) rsum += __shfl_xor(rsum, off, 32);
      l[r] += rsum;
    }

    // O += P @ V  (P re-read from per-wave LDS scratch in A-fragment layout)
    v16bf pa[2];
#pragma unroll
    for (int kk = 0; kk < 2; ++kk) {
      const bf16* p = &Ps[wid][lr * LDP + kk * 32 + hl * 8];
      pa[kk] = mk16(*(const v8bf*)p, *(const v8bf*)(p + 16));
    }
#pragma unroll
    for (int jj = 0; jj < 4; ++jj) {
#pragma unroll
      for (int kk = 0; kk < 2; ++kk) {
        const bf16* p = &Vs[(jj * 16 + lr) * LDV + kk * 32 + hl * 8];
        v16bf vf = mk16(*(const v8bf*)p, *(const v8bf*)(p + 16));
        o[jj] = __builtin_amdgcn_wmma_f32_16x16x32_bf16(
            false, pa[kk], false, vf, (short)0, o[jj], false, false);
      }
    }
  }

  // normalize + merge heads into [b, n, d]
  const int bi = bh >> 4, head = bh & 15;
#pragma unroll
  for (int r = 0; r < 8; ++r) {
    const float inv = 1.0f / l[r];
    const int tok = qr0 + r + hl * 8;
    bf16* dst = O + ((size_t)bi * SEQ + tok) * DIMD + head * DH;
#pragma unroll
    for (int jj = 0; jj < 4; ++jj)
      dst[jj * 16 + lr] = (bf16)(o[jj][r] * inv);
  }
}

// -------- host launch --------
extern "C" void kernel_launch(void* const* d_in, const int* in_sizes, int n_in,
                              void* d_out, int out_size, void* d_ws, size_t ws_size,
                              hipStream_t stream) {
  (void)in_sizes; (void)n_in; (void)out_size; (void)ws_size;
  const float* x   = (const float*)d_in[0];
  const float* Wq  = (const float*)d_in[1];
  const float* bq  = (const float*)d_in[2];
  const float* Wkv = (const float*)d_in[3];
  const float* bkv = (const float*)d_in[4];
  const float* Wo  = (const float*)d_in[5];
  const float* bo  = (const float*)d_in[6];

  char* ws = (char*)d_ws;
  const size_t SZ_X   = (size_t)NTOK * DIMD * 2;        // 16 MiB (also Q/K/Vt/O size)
  const size_t SZ_WQ  = (size_t)DIMD * DIMD * 2;        // 2 MiB
  const size_t SZ_WKV = (size_t)DIMD * 2 * DIMD * 2;    // 4 MiB

  bf16* xb   = (bf16*)(ws);
  bf16* wqb  = (bf16*)(ws + SZ_X);
  bf16* wkvb = (bf16*)(ws + SZ_X + SZ_WQ);
  bf16* wob  = (bf16*)(ws + SZ_X + SZ_WQ + SZ_WKV);
  bf16* Qb   = (bf16*)(ws + SZ_X + SZ_WQ + SZ_WKV + SZ_WQ);
  bf16* Kbuf = (bf16*)((char*)Qb + SZ_X);
  bf16* Vtb  = (bf16*)((char*)Kbuf + SZ_X);
  bf16* Ob   = (bf16*)((char*)Vtb + SZ_X);

  cvt_bf16_kernel<<<2048, 256, 0, stream>>>(x,   xb,   NTOK * DIMD);
  cvt_bf16_kernel<<<512,  256, 0, stream>>>(Wq,  wqb,  DIMD * DIMD);
  cvt_bf16_kernel<<<1024, 256, 0, stream>>>(Wkv, wkvb, DIMD * 2 * DIMD);
  cvt_bf16_kernel<<<512,  256, 0, stream>>>(Wo,  wob,  DIMD * DIMD);

  gemm_wmma_kernel<1><<<dim3(NTOK / BM, DIMD / BN), 256, 0, stream>>>(
      xb, wqb, bq, (void*)Qb, nullptr, DIMD, DIMD);
  gemm_wmma_kernel<2><<<dim3(NTOK / BM, (2 * DIMD) / BN), 256, 0, stream>>>(
      xb, wkvb, bkv, (void*)Kbuf, (void*)Vtb, 2 * DIMD, DIMD);
  flash_attn_kernel<<<dim3(BATCH * HEADS, SEQ / 128), 256, 0, stream>>>(
      Qb, Kbuf, Vtb, Ob);
  gemm_wmma_kernel<0><<<dim3(NTOK / BM, DIMD / BN), 256, 0, stream>>>(
      Ob, wob, bo, d_out, nullptr, DIMD, DIMD);
}
